// SynthesizerAttention_45595372814567
// MI455X (gfx1250) — matrix operand order repaired and checked by reading, not compiled, not run on hardware
//
#include <hip/hip_runtime.h>
#include <hip/hip_bf16.h>
#include <math.h>

#define B_  8
#define S_  2048
#define D_  1024
#define K_  204
#define KP_ 224
#define M_  (B_ * S_)   // 16384

typedef __bf16 bf16;
typedef __attribute__((ext_vector_type(16))) __bf16 v16bf;
typedef __attribute__((ext_vector_type(8)))  __bf16 v8bf;
typedef __attribute__((ext_vector_type(8)))  float  v8f;

static __device__ __forceinline__ v8f vzero8() {
  v8f z = {0.f,0.f,0.f,0.f,0.f,0.f,0.f,0.f};
  return z;
}

// CDNA5 async DMA: global memory -> LDS, tracked with ASYNCcnt.
// lds generic pointer truncates to the wave-relative LDS byte offset.
static __device__ __forceinline__ void async_load_b128(void* lds, const void* gaddr) {
  unsigned l = (unsigned)(size_t)lds;
  asm volatile("global_load_async_to_lds_b128 %0, %1, off"
               :: "v"(l), "v"(gaddr) : "memory");
}
static __device__ __forceinline__ void async_wait0() {
  asm volatile("s_wait_asynccnt 0" ::: "memory");
}

// D = A(16x32 bf16) * B(32x16 bf16) + C(16x16 f32)
// A lane (m = lane&15): elems 0..7  -> K = c0 + half*8 + j ; 8..15 -> +16
// B lane (n = lane&15): elems 0..7  -> K = c0 + half*8 + j ; 8..15 -> +16
// C VGPR r: M = r + 8*half, N = lane&15
static __device__ __forceinline__ v8f wmma_bf16(v8bf alo, v8bf ahi,
                                                v8bf blo, v8bf bhi, v8f c) {
  union { v16bf v; v8bf h[2]; } a, b;
  a.h[0] = alo; a.h[1] = ahi;
  b.h[0] = blo; b.h[1] = bhi;
  return __builtin_amdgcn_wmma_f32_16x16x32_bf16(false, a.v, false, b.v,
                                                 (short)0, c, false, false);
}

// ---------------------------------------------------------------- converts
__global__ __launch_bounds__(256) void cvt_x_kernel(const float* __restrict__ x,
                                                    bf16* __restrict__ xb, int n) {
  int i = (blockIdx.x * 256 + threadIdx.x) * 4;
  if (i >= n) return;
  float4 f = *(const float4*)(x + i);
  xb[i + 0] = (bf16)f.x;
  xb[i + 1] = (bf16)f.y;
  xb[i + 2] = (bf16)f.z;
  xb[i + 3] = (bf16)f.w;
}

// W: [rowsIn][colsIn] f32 row-major  ->  WT: [colsPad][rowsIn] bf16 (zero pad)
__global__ __launch_bounds__(256) void cvt_wT_kernel(const float* __restrict__ W,
                                                     bf16* __restrict__ WT,
                                                     int rowsIn, int colsIn, int colsPad) {
  int idx = blockIdx.x * 256 + threadIdx.x;
  if (idx >= colsPad * rowsIn) return;
  int j = idx / rowsIn;
  int d = idx % rowsIn;
  WT[idx] = (j < colsIn) ? (bf16)W[d * colsIn + j] : (bf16)0.0f;
}

// ---------------------------------------------------------- generic WMMA GEMM
// C[M][N] = A[M][Kd] bf16 * BT[N][Kd]^T bf16 + bias.
// Workgroup tile 256x64, 8 waves; wave w: rows [w*32, w*32+32), 4 N-tiles.
// Double-buffered async (ASYNCcnt) staging of A/B tiles overlaps the WMMAs.
__global__ __launch_bounds__(256) void wmma_gemm_kernel(
    const bf16* __restrict__ A, const bf16* __restrict__ BT,
    const float* __restrict__ bias, int nbias,
    bf16* __restrict__ C, int M, int N, int Kd,
    int lda, int ldb, int ldc, int transposeOut) {
  __shared__ bf16 As[2][256][40];   // pitch 80B: 16B-aligned, bank stride 20
  __shared__ bf16 Bs[2][64][40];

  const int tid  = threadIdx.x;
  const int wave = tid >> 5, lane = tid & 31;
  const int half = lane >> 4, l16 = lane & 15;
  const int m0 = blockIdx.x * 256;
  const int n0 = blockIdx.y * 64;

  const int ar = tid;                               // A: 1 row/thread, 32 elems
  const int br = tid >> 2, bc = (tid & 3) * 8;      // B: 4 thr/row, 8 elems
  const int brow = (n0 + br < N) ? (n0 + br) : (N - 1);  // clamp (tiles discarded)

  v8f acc[2][4] = {{vzero8(), vzero8(), vzero8(), vzero8()},
                   {vzero8(), vzero8(), vzero8(), vzero8()}};

  auto stage = [&](int buf, int k0) {
    const bf16* ap = A + (size_t)(m0 + ar) * lda + k0;
    async_load_b128(&As[buf][ar][0],  ap);
    async_load_b128(&As[buf][ar][8],  ap + 8);
    async_load_b128(&As[buf][ar][16], ap + 16);
    async_load_b128(&As[buf][ar][24], ap + 24);
    async_load_b128(&Bs[buf][br][bc], BT + (size_t)brow * ldb + k0 + bc);
  };

  stage(0, 0);
  for (int k0 = 0; k0 < Kd; k0 += 32) {
    const int cur = (k0 >> 5) & 1;
    async_wait0();        // own staged data complete ...
    __syncthreads();      // ... then everyone's is visible; prev compute done
    if (k0 + 32 < Kd) stage(cur ^ 1, k0 + 32);   // DMA next tile during WMMAs

    const int mrow = wave * 32 + l16;
    v8bf a0lo = *(const v8bf*)&As[cur][mrow][half * 8];
    v8bf a0hi = *(const v8bf*)&As[cur][mrow][16 + half * 8];
    v8bf a1lo = *(const v8bf*)&As[cur][mrow + 16][half * 8];
    v8bf a1hi = *(const v8bf*)&As[cur][mrow + 16][16 + half * 8];
#pragma unroll
    for (int nt = 0; nt < 4; ++nt) {
      const int nrow = nt * 16 + l16;
      v8bf blo = *(const v8bf*)&Bs[cur][nrow][half * 8];
      v8bf bhi = *(const v8bf*)&Bs[cur][nrow][16 + half * 8];
      acc[0][nt] = wmma_bf16(a0lo, a0hi, blo, bhi, acc[0][nt]);
      acc[1][nt] = wmma_bf16(a1lo, a1hi, blo, bhi, acc[1][nt]);
    }
  }

#pragma unroll
  for (int mt = 0; mt < 2; ++mt) {
#pragma unroll
    for (int nt = 0; nt < 4; ++nt) {
      const int col = n0 + nt * 16 + l16;
      if (col >= N) continue;
      const float bv = (col < nbias) ? bias[col] : 0.0f;
#pragma unroll
      for (int r = 0; r < 8; ++r) {
        const int row = m0 + wave * 32 + mt * 16 + half * 8 + r;
        const float v = acc[mt][nt][r] + bv;
        if (!transposeOut) C[(size_t)row * ldc + col] = (bf16)v;
        else               C[(size_t)col * ldc + row] = (bf16)v;
      }
    }
  }
}

// ----------------------------------------------------------- fused attention
// One workgroup: one (batch b, 32-row block). Full score rows live in LDS
// (32 x 2048 f32 = 256 KB -- fits the 320 KB WGP LDS, CDNA5-only).
#define ATTN_SP 2052   // score pitch (f32): 8208B row, bank stride 4
#define ATTN_QP 232    // q pitch (bf16): 464B row (16B-aligned), bank stride 52
#define ATTN_LDS (32 * ATTN_SP * 4 + 32 * ATTN_QP * 2)   // 277,504 B

__global__ __launch_bounds__(256) void attn_kernel(
    const bf16* __restrict__ qb,   // [B*S][KP]
    const bf16* __restrict__ kb,   // [B*S][KP]
    const bf16* __restrict__ vbT,  // [D][B*S]  (v transposed, bias folded in)
    bf16* __restrict__ cb) {       // [B*S][D]
  extern __shared__ char smem[];
  float* sc = (float*)smem;                          // [32][ATTN_SP]
  bf16*  qs = (bf16*)(smem + 32 * ATTN_SP * 4);      // [32][ATTN_QP]

  const int b    = blockIdx.x >> 6;        // 64 row-blocks per batch
  const int m0   = (blockIdx.x & 63) * 32;
  const size_t rowbase = (size_t)b * S_ + m0;

  const int wave = threadIdx.x >> 5, lane = threadIdx.x & 31;
  const int half = lane >> 4, l16 = lane & 15;

  // phase 0: async-DMA q rows [32][KP] into LDS (28 x 16B chunks per row)
  for (int idx = threadIdx.x; idx < 32 * 28; idx += 256) {
    int r = idx / 28, c = (idx % 28) * 8;
    async_load_b128(&qs[r * ATTN_QP + c], &qb[(rowbase + r) * KP_ + c]);
  }
  async_wait0();
  __syncthreads();

  // phase 1: scores = q @ k^T  (256 tile jobs: 128 t-tiles x 2 m-tiles)
  for (int job = wave; job < 256; job += 8) {
    const int mt = job & 1, tt = job >> 1;
    const int arow = mt * 16 + l16;
    const size_t krow = ((size_t)b * S_ + (size_t)(tt * 16 + l16)) * KP_;
    v8f acc = vzero8();
    for (int c0 = 0; c0 < KP_; c0 += 32) {
      v8bf alo = *(const v8bf*)&qs[arow * ATTN_QP + c0 + half * 8];
      v8bf ahi = *(const v8bf*)&qs[arow * ATTN_QP + c0 + 16 + half * 8];
      v8bf blo = *(const v8bf*)&kb[krow + c0 + half * 8];
      v8bf bhi = *(const v8bf*)&kb[krow + c0 + 16 + half * 8];
      acc = wmma_bf16(alo, ahi, blo, bhi, acc);
    }
#pragma unroll
    for (int r = 0; r < 8; ++r)
      sc[(mt * 16 + half * 8 + r) * ATTN_SP + tt * 16 + l16] = acc[r];
  }
  __syncthreads();

  // phase 2: row softmax (wave w owns rows 4w..4w+3; 32-lane reductions)
  for (int rr = 0; rr < 4; ++rr) {
    float* rp = sc + (wave * 4 + rr) * ATTN_SP;
    float mx = -INFINITY;
    for (int t = lane; t < S_; t += 32) mx = fmaxf(mx, rp[t]);
#pragma unroll
    for (int off = 16; off; off >>= 1) mx = fmaxf(mx, __shfl_xor(mx, off, 32));
    float sum = 0.0f;
    for (int t = lane; t < S_; t += 32) { float e = __expf(rp[t] - mx); rp[t] = e; sum += e; }
#pragma unroll
    for (int off = 16; off; off >>= 1) sum += __shfl_xor(sum, off, 32);
    const float inv = 1.0f / sum;
    for (int t = lane; t < S_; t += 32) rp[t] *= inv;
  }
  __syncthreads();

  // phase 3: ctx = probs @ v   (128 jobs: 64 d-tiles x 2 m-tiles)
  for (int job = wave; job < 128; job += 8) {
    const int mt = job & 1, dt = job >> 1;
    const int arow = mt * 16 + l16;
    const size_t vrow = (size_t)(dt * 16 + l16) * (size_t)M_ + (size_t)b * S_;
    v8f acc = vzero8();
    for (int t0 = 0; t0 < S_; t0 += 32) {
      const float* ap0 = &sc[arow * ATTN_SP + t0 + half * 8];
      const float* ap1 = &sc[arow * ATTN_SP + t0 + 16 + half * 8];
      v8bf alo, ahi;
#pragma unroll
      for (int j = 0; j < 8; ++j) { alo[j] = (bf16)ap0[j]; ahi[j] = (bf16)ap1[j]; }
      v8bf blo = *(const v8bf*)&vbT[vrow + t0 + half * 8];
      v8bf bhi = *(const v8bf*)&vbT[vrow + t0 + 16 + half * 8];
      acc = wmma_bf16(alo, ahi, blo, bhi, acc);
    }
    const int d = dt * 16 + l16;
#pragma unroll
    for (int r = 0; r < 8; ++r) {
      const int m = mt * 16 + half * 8 + r;
      cb[(rowbase + m) * D_ + d] = (bf16)acc[r];
    }
  }
}

// ----------------------------- out-projection + residual + LayerNorm (fused)
#define OUT_YP 1028   // y pitch (f32): bank stride 4
#define OUT_CP 1048   // ctx pitch (bf16): 2096B (16B-aligned), bank stride 12
#define OUT_LDS (32 * OUT_YP * 4 + 32 * OUT_CP * 2)   // 198,656 B

__global__ __launch_bounds__(256) void outln_kernel(
    const bf16* __restrict__ cb,   // [M][D]
    const bf16* __restrict__ WdT,  // [D][D] (transposed)
    const float* __restrict__ bd,
    const float* __restrict__ x,   // residual
    const float* __restrict__ gamma, const float* __restrict__ beta,
    float* __restrict__ out) {
  extern __shared__ char smem[];
  float* ys = (float*)smem;                        // [32][OUT_YP]
  bf16*  cs = (bf16*)(smem + 32 * OUT_YP * 4);     // [32][OUT_CP]

  const size_t m0 = (size_t)blockIdx.x * 32;
  const int wave = threadIdx.x >> 5, lane = threadIdx.x & 31;
  const int half = lane >> 4, l16 = lane & 15;

  // stage ctx rows (full K extent) via async DMA: 32 x 128 x 16B chunks
  for (int idx = threadIdx.x; idx < 32 * 128; idx += 256) {
    int r = idx >> 7, c = (idx & 127) * 8;
    async_load_b128(&cs[r * OUT_CP + c], &cb[(m0 + r) * D_ + c]);
  }
  async_wait0();
  __syncthreads();

  // h = ctx @ Wd ; y = h + bd + x    (128 jobs: 64 d-tiles x 2 m-tiles)
  for (int job = wave; job < 128; job += 8) {
    const int mt = job & 1, dt = job >> 1;
    const int arow = mt * 16 + l16;
    const size_t wrow = (size_t)(dt * 16 + l16) * D_;
    v8f acc = vzero8();
    for (int c0 = 0; c0 < D_; c0 += 32) {
      v8bf alo = *(const v8bf*)&cs[arow * OUT_CP + c0 + half * 8];
      v8bf ahi = *(const v8bf*)&cs[arow * OUT_CP + c0 + 16 + half * 8];
      v8bf blo = *(const v8bf*)&WdT[wrow + c0 + half * 8];
      v8bf bhi = *(const v8bf*)&WdT[wrow + c0 + 16 + half * 8];
      acc = wmma_bf16(alo, ahi, blo, bhi, acc);
    }
    const int d = dt * 16 + l16;
    const float bias = bd[d];
#pragma unroll
    for (int r = 0; r < 8; ++r) {
      const int m = mt * 16 + half * 8 + r;
      ys[m * OUT_YP + d] = acc[r] + bias + x[(m0 + m) * D_ + d];
    }
  }
  __syncthreads();

  // LayerNorm: wave w owns rows 4w..4w+3
  for (int rr = 0; rr < 4; ++rr) {
    const int row = wave * 4 + rr;
    const float* rp = ys + row * OUT_YP;
    float s = 0.0f;
    for (int c = lane; c < D_; c += 32) s += rp[c];
#pragma unroll
    for (int off = 16; off; off >>= 1) s += __shfl_xor(s, off, 32);
    const float mean = s * (1.0f / D_);
    float v = 0.0f;
    for (int c = lane; c < D_; c += 32) { float d0 = rp[c] - mean; v += d0 * d0; }
#pragma unroll
    for (int off = 16; off; off >>= 1) v += __shfl_xor(v, off, 32);
    const float rstd = rsqrtf(v * (1.0f / D_) + 1e-12f);
    for (int c = lane; c < D_; c += 32)
      out[(m0 + row) * D_ + c] = gamma[c] * (rp[c] - mean) * rstd + beta[c];
  }
}

// --------------------------------------------------------------------- host
extern "C" void kernel_launch(void* const* d_in, const int* in_sizes, int n_in,
                              void* d_out, int out_size, void* d_ws, size_t ws_size,
                              hipStream_t stream) {
  const float* x     = (const float*)d_in[0];
  const float* Wq    = (const float*)d_in[2];
  const float* bq    = (const float*)d_in[3];
  const float* Wk    = (const float*)d_in[4];
  const float* bk    = (const float*)d_in[5];
  const float* Wv    = (const float*)d_in[6];
  const float* bv    = (const float*)d_in[7];
  const float* Wd    = (const float*)d_in[8];
  const float* bd    = (const float*)d_in[9];
  const float* gamma = (const float*)d_in[10];
  const float* beta  = (const float*)d_in[11];

  char* ws = (char*)d_ws;
  size_t off = 0;
  auto alloc = [&](size_t bytes) { char* p = ws + off; off += (bytes + 255) & ~(size_t)255; return p; };
  bf16* xb  = (bf16*)alloc((size_t)M_ * D_ * 2);
  bf16* qb  = (bf16*)alloc((size_t)M_ * KP_ * 2);
  bf16* kb  = (bf16*)alloc((size_t)M_ * KP_ * 2);
  bf16* vbT = (bf16*)alloc((size_t)D_ * M_ * 2);
  bf16* cb  = (bf16*)alloc((size_t)M_ * D_ * 2);
  bf16* WqT = (bf16*)alloc((size_t)KP_ * D_ * 2);
  bf16* WkT = (bf16*)alloc((size_t)KP_ * D_ * 2);
  bf16* WvT = (bf16*)alloc((size_t)D_ * D_ * 2);
  bf16* WdT = (bf16*)alloc((size_t)D_ * D_ * 2);

  hipFuncSetAttribute((const void*)attn_kernel,
                      hipFuncAttributeMaxDynamicSharedMemorySize, ATTN_LDS);
  hipFuncSetAttribute((const void*)outln_kernel,
                      hipFuncAttributeMaxDynamicSharedMemorySize, OUT_LDS);

  // conversions
  cvt_x_kernel<<<(M_ * D_) / (256 * 4), 256, 0, stream>>>(x, xb, M_ * D_);
  cvt_wT_kernel<<<(KP_ * D_ + 255) / 256, 256, 0, stream>>>(Wq, WqT, D_, K_, KP_);
  cvt_wT_kernel<<<(KP_ * D_ + 255) / 256, 256, 0, stream>>>(Wk, WkT, D_, K_, KP_);
  cvt_wT_kernel<<<(D_ * D_ + 255) / 256, 256, 0, stream>>>(Wv, WvT, D_, D_, D_);
  cvt_wT_kernel<<<(D_ * D_ + 255) / 256, 256, 0, stream>>>(Wd, WdT, D_, D_, D_);

  // projections: q, k (bf16 padded), v (bf16, transposed, bias folded)
  wmma_gemm_kernel<<<dim3(M_ / 256, (KP_ + 63) / 64), 256, 0, stream>>>(
      xb, WqT, bq, K_, qb, M_, KP_, D_, D_, D_, KP_, 0);
  wmma_gemm_kernel<<<dim3(M_ / 256, (KP_ + 63) / 64), 256, 0, stream>>>(
      xb, WkT, bk, K_, kb, M_, KP_, D_, D_, D_, KP_, 0);
  wmma_gemm_kernel<<<dim3(M_ / 256, D_ / 64), 256, 0, stream>>>(
      xb, WvT, bv, D_, vbT, M_, D_, D_, D_, D_, M_, 1);

  // fused scores + softmax + ctx  (one workgroup per 32 query rows)
  attn_kernel<<<dim3(B_ * (S_ / 32)), 256, ATTN_LDS, stream>>>(qb, kb, vbT, cb);

  // fused out-projection + residual + layernorm
  outln_kernel<<<dim3(M_ / 32), 256, OUT_LDS, stream>>>(cb, WdT, bd, x, gamma, beta,
                                                        (float*)d_out);
}
